// FPModule_1915555414492
// MI455X (gfx1250) — matrix operand order repaired and checked by reading, not compiled, not run on hardware
//
#include <hip/hip_runtime.h>
#include <hip/hip_bf16.h>
#include <stdint.h>

typedef __attribute__((ext_vector_type(16))) _Float16 v16h;
typedef __attribute__((ext_vector_type(8)))  _Float16 v8h;
typedef __attribute__((ext_vector_type(4)))  _Float16 v4h;
typedef __attribute__((ext_vector_type(8)))  float    v8f;
typedef __attribute__((ext_vector_type(4)))  float    v4f;

#define NB    8
#define NC    1024
#define NF    4096
#define CIN   256
#define CSK   128
#define NOUT  256
#define MTOT  (NB * NF)          // 32768 rows
#define DIN   (CIN + CSK)        // 384

// ---------------------------------------------------------------------------
// CDNA5 async global->LDS copy (16B per lane, tracked by ASYNCcnt) and wait.
// LDS byte address taken from a VGPR (low 32 bits of the generic pointer = LDS offset).
// ---------------------------------------------------------------------------
__device__ __forceinline__ uint32_t lds_addr32(const void* p) {
    return (uint32_t)(uintptr_t)p;
}
__device__ __forceinline__ void async_copy_b128(uint32_t lds_byte_addr, const void* gptr) {
    asm volatile("global_load_async_to_lds_b128 %0, %1, off"
                 :: "v"(lds_byte_addr), "v"(gptr) : "memory");
}
__device__ __forceinline__ void wait_async0() {
    asm volatile("s_wait_asynccnt 0x0" ::: "memory");
}

// ---------------------------------------------------------------------------
// Kernel 1: per fine point, 3-NN over its batch's coarse points (LDS cached)
// and normalized inverse-d^2 weights.
// ---------------------------------------------------------------------------
__global__ __launch_bounds__(256) void knn_kernel(const float* __restrict__ pos,
                                                  const float* __restrict__ pos_skip,
                                                  float* __restrict__ knn_w,
                                                  int*   __restrict__ knn_idx) {
    __shared__ float spx[NC], spy[NC], spz[NC];
    const int t = threadIdx.x;
    const int b = blockIdx.x >> 4;              // 16 blocks of 256 per batch
    const int j = blockIdx.x * 256 + t;
    const float* pb = pos + (size_t)b * NC * 3;
    for (int i = t; i < NC; i += 256) {
        spx[i] = pb[i * 3 + 0];
        spy[i] = pb[i * 3 + 1];
        spz[i] = pb[i * 3 + 2];
    }
    __syncthreads();

    const float qx = pos_skip[j * 3 + 0];
    const float qy = pos_skip[j * 3 + 1];
    const float qz = pos_skip[j * 3 + 2];

    float d0 = 3.4e38f, d1 = 3.4e38f, d2 = 3.4e38f;
    int   i0 = 0, i1 = 0, i2 = 0;
#pragma unroll 4
    for (int i = 0; i < NC; ++i) {
        const float dx = qx - spx[i];
        const float dy = qy - spy[i];
        const float dz = qz - spz[i];
        const float d  = dx * dx + dy * dy + dz * dz;
        if (d < d2) {
            if (d < d1) {
                d2 = d1; i2 = i1;
                if (d < d0) { d1 = d0; i1 = i0; d0 = d; i0 = i; }
                else        { d1 = d;  i1 = i; }
            } else { d2 = d; i2 = i; }
        }
    }
    float w0 = 1.0f / fmaxf(d0, 1e-16f);
    float w1 = 1.0f / fmaxf(d1, 1e-16f);
    float w2 = 1.0f / fmaxf(d2, 1e-16f);
    const float inv = 1.0f / (w0 + w1 + w2);
    knn_w[j * 3 + 0] = w0 * inv;
    knn_w[j * 3 + 1] = w1 * inv;
    knn_w[j * 3 + 2] = w2 * inv;
    knn_idx[j * 3 + 0] = i0;
    knn_idx[j * 3 + 1] = i1;
    knn_idx[j * 3 + 2] = i2;
}

// ---------------------------------------------------------------------------
// Kernel 2: build z = [knn-interp(x) | x_skip] (32768 x 384) directly as
// hi/lo split f16 pairs -- one conversion per element, ready for async DMA.
// Thread handles one float4: col = 4*q, q in [0,96): q<64 -> interp, else skip.
// ---------------------------------------------------------------------------
__global__ __launch_bounds__(256) void build_z_kernel(const float* __restrict__ x,
                                                      const float* __restrict__ x_skip,
                                                      const float* __restrict__ knn_w,
                                                      const int*   __restrict__ knn_idx,
                                                      _Float16* __restrict__ zh,
                                                      _Float16* __restrict__ zl) {
    const long long tid = (long long)blockIdx.x * 256 + threadIdx.x;   // MTOT*96
    const int j = (int)(tid / 96);
    const int q = (int)(tid % 96);
    const int col = q << 2;                       // 0..380, col>=256 -> skip part
    v4f v;
    if (q < 64) {
        const int b  = j >> 12;
        const int i0 = knn_idx[j * 3 + 0];
        const int i1 = knn_idx[j * 3 + 1];
        const int i2 = knn_idx[j * 3 + 2];
        const float w0 = knn_w[j * 3 + 0];
        const float w1 = knn_w[j * 3 + 1];
        const float w2 = knn_w[j * 3 + 2];
        const float* xb = x + (size_t)b * NC * CIN;
        const v4f f0 = *(const v4f*)(xb + (size_t)i0 * CIN + col);
        const v4f f1 = *(const v4f*)(xb + (size_t)i1 * CIN + col);
        const v4f f2 = *(const v4f*)(xb + (size_t)i2 * CIN + col);
        v = f0 * w0 + f1 * w1 + f2 * w2;
    } else {
        v = *(const v4f*)(x_skip + (size_t)j * CSK + (col - CIN));
    }
    v4h hv, lv;
#pragma unroll
    for (int i = 0; i < 4; ++i) {
        const _Float16 h = (_Float16)v[i];
        hv[i] = h;
        lv[i] = (_Float16)(v[i] - (float)h);
    }
    *(v4h*)(zh + (size_t)j * DIN + col) = hv;
    *(v4h*)(zl + (size_t)j * DIN + col) = lv;
}

// ---------------------------------------------------------------------------
// Kernel 3: W (Kdim x 256 row-major f32) -> column-major hi/lo f16 (256 x Kdim)
// so B tiles become contiguous per column for async b128 staging.
// ---------------------------------------------------------------------------
__global__ __launch_bounds__(256) void convert_wt_kernel(const float* __restrict__ W,
                                                         _Float16* __restrict__ Wth,
                                                         _Float16* __restrict__ Wtl,
                                                         int Kdim) {
    const int k = blockIdx.x;
    const int n = threadIdx.x;
    const float f = W[(size_t)k * 256 + n];
    const _Float16 h = (_Float16)f;
    Wth[(size_t)n * Kdim + k] = h;
    Wtl[(size_t)n * Kdim + k] = (_Float16)(f - (float)h);
}

// ---------------------------------------------------------------------------
// WMMA helpers: fp32 emulated as hi/lo f16 split, f32 accumulate.
// ---------------------------------------------------------------------------
__device__ __forceinline__ v16h cat8(v8h a, v8h b) {
    return __builtin_shufflevector(a, b, 0, 1, 2, 3, 4, 5, 6, 7,
                                         8, 9, 10, 11, 12, 13, 14, 15);
}
__device__ __forceinline__ v8f wmma16(v16h a, v16h b, v8f c) {
    return __builtin_amdgcn_wmma_f32_16x16x32_f16(false, a, false, b,
                                                  (short)0, c, false, false);
}

// ---------------------------------------------------------------------------
// Kernel 4/5: out = relu(A @ W + bias), A/W pre-split hi/lo f16.
// Block = 256 threads (8 wave32), tile 64(M) x 128(N), K-step 32.
// A/B tiles staged by async global->LDS DMA, double-buffered:
//   issue(k+1) -> wmma-compute(k) -> s_wait_asynccnt 0 -> barrier.
// Epilogue: fp32 store (final layer) or hi/lo f16 store (hidden layer).
// ---------------------------------------------------------------------------
__global__ __launch_bounds__(256) void mlp_gemm_kernel(const _Float16* __restrict__ Ah,
                                                       const _Float16* __restrict__ Al,
                                                       const _Float16* __restrict__ Bth,
                                                       const _Float16* __restrict__ Btl,
                                                       const float* __restrict__ bias,
                                                       float* __restrict__ outF,
                                                       _Float16* __restrict__ outH,
                                                       _Float16* __restrict__ outL,
                                                       int Kdim) {
    __shared__ __align__(16) _Float16 AsH[2][64 * 32];
    __shared__ __align__(16) _Float16 AsL[2][64 * 32];
    __shared__ __align__(16) _Float16 BsH[2][128 * 32];
    __shared__ __align__(16) _Float16 BsL[2][128 * 32];   // 48 KB total

    const int t    = threadIdx.x;
    const int m0   = (blockIdx.x >> 1) * 64;
    const int n0   = (blockIdx.x & 1) * 128;
    const int wave = t >> 5;
    const int lane = t & 31;
    const int wm   = wave & 3;      // M sub-tile 0..3
    const int wn   = wave >> 2;     // N half 0..1 (4 sub-tiles of 16 each)
    const int hh   = lane >> 4;
    const int r    = lane & 15;

    // Per-thread async-copy assignment (constant across K iterations):
    const int a_row = t >> 2;            // A: 64 rows x 4 chunks of 16B
    const int a_off = (t & 3) * 8;       // in halves
    const _Float16* gAh = Ah + (size_t)(m0 + a_row) * Kdim + a_off;
    const _Float16* gAl = Al + (size_t)(m0 + a_row) * Kdim + a_off;

    v8f acc[4] = {};

    auto stage = [&](int kt, int buf) {
        const int k0 = kt << 5;
        async_copy_b128(lds_addr32(&AsH[buf][a_row * 32 + a_off]), gAh + k0);
        async_copy_b128(lds_addr32(&AsL[buf][a_row * 32 + a_off]), gAl + k0);
#pragma unroll
        for (int q = 0; q < 2; ++q) {        // B: 128 cols x 4 chunks, hi+lo
            const int c   = q * 256 + t;
            const int n   = c >> 2;
            const int off = (c & 3) * 8;
            async_copy_b128(lds_addr32(&BsH[buf][n * 32 + off]),
                            Bth + (size_t)(n0 + n) * Kdim + k0 + off);
            async_copy_b128(lds_addr32(&BsL[buf][n * 32 + off]),
                            Btl + (size_t)(n0 + n) * Kdim + k0 + off);
        }
    };

    const int nK = Kdim >> 5;
    stage(0, 0);
    wait_async0();
    __syncthreads();

    for (int kt = 0; kt < nK; ++kt) {
        const int buf = kt & 1;
        if (kt + 1 < nK) stage(kt + 1, buf ^ 1);   // DMA overlaps WMMA below

        // A fragments (documented 16-bit A layout: half hh holds K in
        // {8hh..8hh+7, 16+8hh..16+8hh+7})
        const int arow = (wm * 16 + r) * 32;
        const v16h aH = cat8(*(const v8h*)&AsH[buf][arow + 8 * hh],
                             *(const v8h*)&AsH[buf][arow + 16 + 8 * hh]);
        const v16h aL = cat8(*(const v8h*)&AsL[buf][arow + 8 * hh],
                             *(const v8h*)&AsL[buf][arow + 16 + 8 * hh]);
#pragma unroll
        for (int s = 0; s < 4; ++s) {
            const int ncol = (wn * 64 + s * 16 + r) * 32 + 8 * hh;
            const v16h bH = cat8(*(const v8h*)&BsH[buf][ncol],
                                 *(const v8h*)&BsH[buf][ncol + 16]);
            const v16h bL = cat8(*(const v8h*)&BsL[buf][ncol],
                                 *(const v8h*)&BsL[buf][ncol + 16]);
            acc[s] = wmma16(aH, bH, acc[s]);
            acc[s] = wmma16(aH, bL, acc[s]);
            acc[s] = wmma16(aL, bH, acc[s]);
        }
        wait_async0();
        __syncthreads();
    }

    // Epilogue. C/D layout: lane<16 -> M=v, N=lane; lane>=16 -> M=v+8, N=lane-16
#pragma unroll
    for (int s = 0; s < 4; ++s) {
        const int col = n0 + wn * 64 + s * 16 + r;
        const float bv = bias[col];
#pragma unroll
        for (int v = 0; v < 8; ++v) {
            const int row = m0 + wm * 16 + v + 8 * hh;
            const float o = fmaxf(acc[s][v] + bv, 0.0f);
            if (outF) {
                outF[(size_t)row * 256 + col] = o;
            } else {
                const _Float16 h = (_Float16)o;
                outH[(size_t)row * 256 + col] = h;
                outL[(size_t)row * 256 + col] = (_Float16)(o - (float)h);
            }
        }
    }
}

// ---------------------------------------------------------------------------
// Kernel 6: tuple tail: pos_skip and batch_skip copied into d_out.
// ---------------------------------------------------------------------------
__global__ __launch_bounds__(256) void tail_kernel(const float* __restrict__ pos_skip,
                                                   const long long* __restrict__ batch_skip,
                                                   float* __restrict__ out_tail) {
    const int i = blockIdx.x * 256 + threadIdx.x;
    if (i < MTOT * 3) out_tail[i] = pos_skip[i];
    if (i < MTOT)     out_tail[MTOT * 3 + i] = (float)batch_skip[i];
}

// ---------------------------------------------------------------------------
extern "C" void kernel_launch(void* const* d_in, const int* in_sizes, int n_in,
                              void* d_out, int out_size, void* d_ws, size_t ws_size,
                              hipStream_t stream) {
    const float*      x          = (const float*)d_in[0];
    const float*      pos        = (const float*)d_in[1];
    /* d_in[2] = batch (unused) */
    const float*      x_skip     = (const float*)d_in[3];
    const float*      pos_skip   = (const float*)d_in[4];
    const long long*  batch_skip = (const long long*)d_in[5];
    const float*      W1         = (const float*)d_in[6];
    const float*      b1         = (const float*)d_in[7];
    const float*      W2         = (const float*)d_in[8];
    const float*      b2         = (const float*)d_in[9];
    float*            out        = (float*)d_out;

    float*    knn_w   = (float*)d_ws;                         // 32768*3 f32
    int*      knn_idx = (int*)(knn_w + (size_t)MTOT * 3);     // 32768*3 i32
    _Float16* zh      = (_Float16*)(knn_idx + (size_t)MTOT * 3);
    _Float16* zl      = zh  + (size_t)MTOT * DIN;             // 2 x 24 MB
    _Float16* hbh     = zl  + (size_t)MTOT * DIN;
    _Float16* hbl     = hbh + (size_t)MTOT * 256;             // 2 x 16 MB
    _Float16* w1th    = hbl + (size_t)MTOT * 256;
    _Float16* w1tl    = w1th + (size_t)DIN * 256;
    _Float16* w2th    = w1tl + (size_t)DIN * 256;
    _Float16* w2tl    = w2th + (size_t)256 * 256;

    knn_kernel    <<<MTOT / 256, 256, 0, stream>>>(pos, pos_skip, knn_w, knn_idx);
    build_z_kernel<<<MTOT * 96 / 256, 256, 0, stream>>>(x, x_skip, knn_w, knn_idx, zh, zl);
    convert_wt_kernel<<<DIN, 256, 0, stream>>>(W1, w1th, w1tl, DIN);
    convert_wt_kernel<<<256, 256, 0, stream>>>(W2, w2th, w2tl, 256);
    // GEMM1: z (384) -> h (256), relu, h written as hi/lo f16
    mlp_gemm_kernel<<<(MTOT / 64) * 2, 256, 0, stream>>>(zh, zl, w1th, w1tl, b1,
                                                         nullptr, hbh, hbl, DIN);
    // GEMM2: h (256) -> out (256), relu, fp32 output
    mlp_gemm_kernel<<<(MTOT / 64) * 2, 256, 0, stream>>>(hbh, hbl, w2th, w2tl, b2,
                                                         out, nullptr, nullptr, 256);
    tail_kernel<<<MTOT * 3 / 256, 256, 0, stream>>>(pos_skip, batch_skip,
                                                    out + (size_t)MTOT * NOUT);
}
